// SceneGraphformerattention_5927054868928
// MI455X (gfx1250) — compile-verified, matrix operand order checked by reading, not run on hardware
//
#include <hip/hip_runtime.h>
#include <hip/hip_bf16.h>

// ---------------------------------------------------------------------------
// CDNA5 (gfx1250) implementation.
//  - All heavy GEMMs + attention on v_wmma_f32_16x16x32_f16 (wave32 WMMA).
//  - Weights pre-converted once per call to transposed f16 [N][K] (padded),
//    then streamed into LDS with global_load_async_to_lds_b128 (ASYNCcnt).
//  - Fragment loads are ds_load_b128 (fragment-contiguous LDS layouts).
//  - Graph scatter/softmax use atomics.
// ---------------------------------------------------------------------------

typedef __attribute__((ext_vector_type(16))) _Float16 v16h;
typedef __attribute__((ext_vector_type(8)))  _Float16 v8h;
typedef __attribute__((ext_vector_type(4)))  _Float16 v4h;
typedef __attribute__((ext_vector_type(8)))  float    v8f;

constexpr int cB = 128, cS = 128, cH = 512, cL = 6, cFF = 2048, cNH = 8, cHD = 64;
constexpr int cN = 12288, cIN = 512, cE = 196608, cEE = cE + cN;
constexpr int cFLAT = 1024, cNCLS = 3129, cNCLSP = 3136;  // padded to 64
constexpr int cBS = cB * cS;

#define SHUF16(lo, hi) __builtin_shufflevector(lo, hi, 0, 1, 2, 3, 4, 5, 6, 7, \
                                               8, 9, 10, 11, 12, 13, 14, 15)

// ---------------------------------------------------------------------------
// helpers
// ---------------------------------------------------------------------------
__device__ __forceinline__ float block_reduce_sum(float v, float* sh) {
  int t = threadIdx.x;
  sh[t] = v;
  __syncthreads();
  for (int s = blockDim.x >> 1; s > 0; s >>= 1) {
    if (t < s) sh[t] += sh[t + s];
    __syncthreads();
  }
  float r = sh[0];
  __syncthreads();
  return r;
}

__global__ void zero_kernel(float* p, size_t n) {
  size_t i = (size_t)blockIdx.x * blockDim.x + threadIdx.x;
  size_t st = (size_t)gridDim.x * blockDim.x;
  for (; i < n; i += st) p[i] = 0.f;
}

// convert W[K,N] f32 -> Wt[Npad][K] f16 (transposed, zero-padded rows)
__global__ void wconv_kernel(const float* __restrict__ Wsrc,
                             _Float16* __restrict__ Wt,
                             int K, int N, int Npad) {
  size_t total = (size_t)Npad * K;
  for (size_t i = (size_t)blockIdx.x * blockDim.x + threadIdx.x; i < total;
       i += (size_t)gridDim.x * blockDim.x) {
    int n = (int)(i / K), k = (int)(i % K);
    Wt[i] = (n < N) ? (_Float16)Wsrc[(size_t)k * N + n] : (_Float16)0.f;
  }
}

// ---------------------------------------------------------------------------
// WMMA GEMM: C[M,N] = act(A[M,K] @ W[K,N] + bias), W given as Bt[Npad][K] f16.
// block = 256 threads = 8 wave32; block tile 64(M) x 64(N); K step 32.
// B tile (4KB) staged via async load-to-LDS; A tile staged with cvt_pk.
// Each wave owns two 16x16 D tiles (shared B fragment) -> 2 WMMA / K-step.
// Requires K % 32 == 0 and Bt rows padded to N%64==0 (both hold here).
// ---------------------------------------------------------------------------
__global__ void wmma_gemm_kernel(const float* __restrict__ A, int lda,
                                 const _Float16* __restrict__ Bt,
                                 const float* __restrict__ bias,
                                 float* __restrict__ C, int ldc,
                                 int M, int N, int K, int act) {
  // single shared block (static LDS base 0): As[64][40] then Bst[64][40]
  __shared__ __align__(16) char smem[2 * 64 * 40 * 2];
  _Float16 (*As)[40]  = reinterpret_cast<_Float16(*)[40]>(smem);
  _Float16 (*Bst)[40] = reinterpret_cast<_Float16(*)[40]>(smem + 64 * 40 * 2);
  constexpr unsigned BST_OFF = 64 * 40 * 2;     // LDS byte offset of Bst

  const int tid  = threadIdx.x;
  const int lane = tid & 31;
  const int wave = tid >> 5;          // 0..7
  const int wm = wave & 1;            // wave covers rows [wm*32, wm*32+32)
  const int wn = wave >> 1;           // wave covers cols [wn*16, wn*16+16)
  const int bm = blockIdx.y * 64;
  const int bn = blockIdx.x * 64;
  const int hl = lane >> 4;           // half-wave select
  const int l15 = lane & 15;
  const int khalf = hl * 8;

  v8f c0 = {0.f, 0.f, 0.f, 0.f, 0.f, 0.f, 0.f, 0.f};
  v8f c1 = {0.f, 0.f, 0.f, 0.f, 0.f, 0.f, 0.f, 0.f};

  const int br = tid >> 2;            // B-tile row (n) this thread copies
  const int bq = tid & 3;             // 16B chunk within row

  for (int k0 = 0; k0 < K; k0 += 32) {
    // ---- async-stage B: 64(n) x 32(k) f16 tile = 4KB, 16B per thread ----
    {
      const _Float16* g = &Bt[(size_t)(bn + br) * K + (k0 + bq * 8)];
      unsigned dst = BST_OFF + (unsigned)(br * 80 + bq * 16);
      unsigned long long ga = (unsigned long long)g;
      asm volatile("global_load_async_to_lds_b128 %0, %1, off"
                   :: "v"(dst), "v"(ga) : "memory");
    }
    // ---- stage A: 64x32 tile, float4 loads, packed f16 b64 stores ----
#pragma unroll
    for (int t2 = 0; t2 < 2; ++t2) {
      int idx = (tid + t2 * 256) << 2;          // element index, step 4
      int r = idx >> 5, kk = idx & 31;
      int gm = bm + r, gk = k0 + kk;
      v4h hv = {(_Float16)0.f, (_Float16)0.f, (_Float16)0.f, (_Float16)0.f};
      if (gm < M) {
        float4 v = *(const float4*)&A[(size_t)gm * lda + gk];
        hv[0] = (_Float16)v.x; hv[1] = (_Float16)v.y;
        hv[2] = (_Float16)v.z; hv[3] = (_Float16)v.w;
      }
      *(v4h*)&As[r][kk] = hv;
    }
    if (k0 + 32 < K) {      // prefetch next A tile (global_prefetch path)
      int r = tid >> 2;
      if (bm + r < M) __builtin_prefetch(&A[(size_t)(bm + r) * lda + k0 + 32], 0, 1);
    }
    asm volatile("s_wait_asynccnt 0x0" ::: "memory");
    __syncthreads();

    // ---- fragments: two ds_load_b128 each + register shuffle ----
    v8h blo = *(const v8h*)&Bst[wn * 16 + l15][khalf];
    v8h bhi = *(const v8h*)&Bst[wn * 16 + l15][khalf + 16];
    v16h bf = SHUF16(blo, bhi);

    v8h alo0 = *(const v8h*)&As[wm * 32 + l15][khalf];
    v8h ahi0 = *(const v8h*)&As[wm * 32 + l15][khalf + 16];
    v16h a0 = SHUF16(alo0, ahi0);
    c0 = __builtin_amdgcn_wmma_f32_16x16x32_f16(false, a0, false, bf,
                                                (short)0, c0, false, false);

    v8h alo1 = *(const v8h*)&As[wm * 32 + 16 + l15][khalf];
    v8h ahi1 = *(const v8h*)&As[wm * 32 + 16 + l15][khalf + 16];
    v16h a1 = SHUF16(alo1, ahi1);
    c1 = __builtin_amdgcn_wmma_f32_16x16x32_f16(false, a1, false, bf,
                                                (short)0, c1, false, false);
    __syncthreads();
  }

  const int col = bn + wn * 16 + l15;
  if (col < N) {
    float bv = bias ? bias[col] : 0.f;
#pragma unroll
    for (int r = 0; r < 8; ++r) {
      int row0 = bm + wm * 32 + hl * 8 + r;
      if (row0 < M) {
        float v = c0[r] + bv;
        if (act == 1) v = v > 0.f ? v : 0.f;
        C[(size_t)row0 * ldc + col] = v;
      }
      int row1 = row0 + 16;
      if (row1 < M) {
        float v = c1[r] + bv;
        if (act == 1) v = v > 0.f ? v : 0.f;
        C[(size_t)row1 * ldc + col] = v;
      }
    }
  }
}

// ---------------------------------------------------------------------------
// Fused attention: per (batch, head, 32-query tile). S=128, HD=64, stride H.
// scores = Q K^T / 8 (masked), softmax, O = P V.  All matmuls on WMMA.
// Score f32 buffer is overlaid on the dead K tile -> 48KB static LDS total.
// ---------------------------------------------------------------------------
__global__ void attention_kernel(const float* __restrict__ Q,
                                 const float* __restrict__ Kv,
                                 const float* __restrict__ V,
                                 const int* __restrict__ maskKeys,  // [B*S] 1=masked
                                 float* __restrict__ O) {
  __shared__ __align__(16) _Float16 Qs[32][72];      // [q][k]   4.6KB
  __shared__ __align__(16) _Float16 KsS[128][72];    // [n][k] -> later f32 Ss[32][129]
  __shared__ __align__(16) _Float16 Vst[64][136];    // [n][k]  17.4KB
  __shared__ __align__(16) _Float16 Ps[32][136];     // [q][k]   8.7KB

  const int tid = threadIdx.x;
  const int lane = tid & 31;
  const int wave = tid >> 5;
  const int qt = blockIdx.x;
  const int hh = blockIdx.y;
  const int bz = blockIdx.z;
  const size_t base = (size_t)bz * cS * cH + (size_t)hh * cHD;
  const int q0 = qt * 32;
  const int hl = lane >> 4, l15 = lane & 15, khalf = hl * 8;

  // stage K (row-major) and V (transposed)
  for (int i = tid; i < 2048; i += 256) {            // 128x64 / 4
    int s = i >> 4;
    int d4 = (i & 15) << 2;
    size_t o = base + (size_t)s * cH + d4;
    float4 kv = *(const float4*)&Kv[o];
    v4h kh;
    kh[0] = (_Float16)kv.x; kh[1] = (_Float16)kv.y;
    kh[2] = (_Float16)kv.z; kh[3] = (_Float16)kv.w;
    *(v4h*)&KsS[s][d4] = kh;
    float4 vv = *(const float4*)&V[o];
    Vst[d4 + 0][s] = (_Float16)vv.x;
    Vst[d4 + 1][s] = (_Float16)vv.y;
    Vst[d4 + 2][s] = (_Float16)vv.z;
    Vst[d4 + 3][s] = (_Float16)vv.w;
  }
  for (int i = tid; i < 512; i += 256) {             // 32x64 / 4
    int r = i >> 4;
    int d4 = (i & 15) << 2;
    float4 qv = *(const float4*)&Q[base + (size_t)(q0 + r) * cH + d4];
    v4h qh;
    qh[0] = (_Float16)qv.x; qh[1] = (_Float16)qv.y;
    qh[2] = (_Float16)qv.z; qh[3] = (_Float16)qv.w;
    *(v4h*)&Qs[r][d4] = qh;
  }
  __syncthreads();

  // ---- scores: 2(m) x 8(n) tiles of 16x16, 2 per wave, K=64 (2 wmma steps)
  v8f acc[2];
#pragma unroll
  for (int ti = 0; ti < 2; ++ti) {
    int t = wave + ti * 8;
    int tm = t & 1, tn = t >> 1;
    v8f c = {0.f, 0.f, 0.f, 0.f, 0.f, 0.f, 0.f, 0.f};
#pragma unroll
    for (int ks = 0; ks < 2; ++ks) {
      int ko = ks * 32 + khalf;
      v8h alo = *(const v8h*)&Qs[tm * 16 + l15][ko];
      v8h ahi = *(const v8h*)&Qs[tm * 16 + l15][ko + 16];
      v8h blo = *(const v8h*)&KsS[tn * 16 + l15][ko];   // B = K^T: row read
      v8h bhi = *(const v8h*)&KsS[tn * 16 + l15][ko + 16];
      v16h a = SHUF16(alo, ahi);
      v16h bf = SHUF16(blo, bhi);
      c = __builtin_amdgcn_wmma_f32_16x16x32_f16(false, a, false, bf,
                                                 (short)0, c, false, false);
    }
    acc[ti] = c;
  }
  __syncthreads();     // all K reads done -> safe to overlay Ss on KsS

  float (*Ss)[129] = reinterpret_cast<float(*)[129]>(&KsS[0][0]);
#pragma unroll
  for (int ti = 0; ti < 2; ++ti) {
    int t = wave + ti * 8;
    int tm = t & 1, tn = t >> 1;
    int n = tn * 16 + l15;
    int masked = maskKeys[bz * cS + n];
#pragma unroll
    for (int r = 0; r < 8; ++r) {
      int m = tm * 16 + hl * 8 + r;
      Ss[m][n] = masked ? -1.0e9f : acc[ti][r] * 0.125f;
    }
  }
  __syncthreads();

  // ---- row softmax (32 rows), one row per thread ----
  if (tid < 32) {
    float mx = -3.4e38f;
    for (int n = 0; n < cS; ++n) mx = fmaxf(mx, Ss[tid][n]);
    float sum = 0.f;
    for (int n = 0; n < cS; ++n) {
      float e = __expf(Ss[tid][n] - mx);
      Ss[tid][n] = e;
      sum += e;
    }
    float inv = 1.f / sum;
    for (int n = 0; n < cS; ++n) Ps[tid][n] = (_Float16)(Ss[tid][n] * inv);
  }
  __syncthreads();

  // ---- O = P(32x128) @ V(128x64): 2(m) x 4(n) tiles, one per wave, K=128
  {
    int tm = wave & 1, tn = wave >> 1;
    v8f c = {0.f, 0.f, 0.f, 0.f, 0.f, 0.f, 0.f, 0.f};
#pragma unroll
    for (int ks = 0; ks < 4; ++ks) {
      int ko = ks * 32 + khalf;
      v8h alo = *(const v8h*)&Ps[tm * 16 + l15][ko];
      v8h ahi = *(const v8h*)&Ps[tm * 16 + l15][ko + 16];
      v8h blo = *(const v8h*)&Vst[tn * 16 + l15][ko];
      v8h bhi = *(const v8h*)&Vst[tn * 16 + l15][ko + 16];
      v16h a = SHUF16(alo, ahi);
      v16h bf = SHUF16(blo, bhi);
      c = __builtin_amdgcn_wmma_f32_16x16x32_f16(false, a, false, bf,
                                                 (short)0, c, false, false);
    }
    int n = tn * 16 + l15;
#pragma unroll
    for (int r = 0; r < 8; ++r) {
      int m = tm * 16 + hl * 8 + r;
      O[base + (size_t)(q0 + m) * cH + n] = c[r];
    }
  }
}

// ---------------------------------------------------------------------------
// GAT / graph kernels
// ---------------------------------------------------------------------------
__global__ void rowdot2_kernel(const float* __restrict__ h,
                               const float* __restrict__ wa,
                               const float* __restrict__ wb,
                               float* oa, float* ob, int D) {
  __shared__ float sh[256];
  int row = blockIdx.x, t = threadIdx.x;
  float s1 = 0.f, s2 = 0.f;
  for (int c = t; c < D; c += 256) {
    float v = h[(size_t)row * D + c];
    s1 += v * wa[c];
    s2 += v * wb[c];
  }
  float r1 = block_reduce_sum(s1, sh);
  float r2 = block_reduce_sum(s2, sh);
  if (t == 0) { oa[row] = r1; ob[row] = r2; }
}

__device__ __forceinline__ unsigned enc_f(float f) {
  unsigned u = __float_as_uint(f);
  return (u & 0x80000000u) ? ~u : (u | 0x80000000u);
}
__device__ __forceinline__ float dec_f(unsigned u) {
  return (u & 0x80000000u) ? __uint_as_float(u & 0x7FFFFFFFu)
                           : __uint_as_float(~u);
}

__global__ void edge_e_max_kernel(const int* __restrict__ ei,
                                  const float* __restrict__ hs,
                                  const float* __restrict__ hd,
                                  float* e_buf, unsigned* emax) {
  int i = blockIdx.x * blockDim.x + threadIdx.x;
  if (i >= cEE) return;
  int s = (i < cE) ? ei[i]       : (i - cE);
  int d = (i < cE) ? ei[cE + i]  : (i - cE);
  float v = hs[s] + hd[d];
  v = v > 0.f ? v : 0.2f * v;            // leaky relu
  e_buf[i] = v;
  atomicMax(&emax[d], enc_f(v));
}

__global__ void edge_softmax_kernel(const int* __restrict__ ei,
                                    const float* __restrict__ e_buf,
                                    const unsigned* __restrict__ emax,
                                    float* ee, float* den) {
  int i = blockIdx.x * blockDim.x + threadIdx.x;
  if (i >= cEE) return;
  int d = (i < cE) ? ei[cE + i] : (i - cE);
  float x = __expf(e_buf[i] - dec_f(emax[d]));
  ee[i] = x;
  atomicAdd(&den[d], x);
}

__global__ void edge_alpha_kernel(const int* __restrict__ ei,
                                  const float* __restrict__ ee,
                                  const float* __restrict__ den,
                                  float* alpha, float* alpha_out) {
  int i = blockIdx.x * blockDim.x + threadIdx.x;
  if (i >= cEE) return;
  int d = (i < cE) ? ei[cE + i] : (i - cE);
  float a = ee[i] / (den[d] + 1e-16f);
  alpha[i] = a;
  alpha_out[i] = a;
}

__global__ void scatter_feat_kernel(const int* __restrict__ ei,
                                    const float* __restrict__ h,
                                    const float* __restrict__ alpha,
                                    float* xacc) {
  int edge = blockIdx.x;
  int s = (edge < cE) ? ei[edge]      : (edge - cE);
  int d = (edge < cE) ? ei[cE + edge] : (edge - cE);
  float a = alpha[edge];
  for (int c = threadIdx.x; c < cH; c += blockDim.x)
    atomicAdd(&xacc[(size_t)d * cH + c], h[(size_t)s * cH + c] * a);
}

__global__ void bn_stats_kernel(const float* __restrict__ x,
                                const float* __restrict__ gb,
                                float* mu, float* var) {
  __shared__ float sh[256];
  int c = blockIdx.x, t = threadIdx.x;
  float b0 = gb[c];
  float s = 0.f, ss = 0.f;
  for (int i = t; i < cN; i += 256) {
    float v = x[(size_t)i * cH + c] + b0;
    s += v;
    ss += v * v;
  }
  float ts = block_reduce_sum(s, sh);
  float tss = block_reduce_sum(ss, sh);
  if (t == 0) {
    float m = ts / cN;
    mu[c] = m;
    var[c] = tss / cN - m * m;
  }
}

__global__ void count_labels_kernel(const int* lab, int* counts) {
  int i = blockIdx.x * blockDim.x + threadIdx.x;
  if (i < cN) atomicAdd(&counts[lab[i]], 1);
}
__global__ void scan_offs_kernel(const int* counts, int* offs) {
  if (threadIdx.x == 0 && blockIdx.x == 0) {
    int a = 0;
    for (int i = 0; i < cB; ++i) { offs[i] = a; a += counts[i]; }
  }
}
__global__ void compute_pos_kernel(const int* lab, const int* offs, int* pos) {
  int i = blockIdx.x * blockDim.x + threadIdx.x;
  if (i < cN) pos[i] = i - offs[lab[i]];
}

__global__ void init_masks_kernel(const int* tok, int* m_self, int* m_cross) {
  int i = blockIdx.x * blockDim.x + threadIdx.x;
  if (i < cBS) {
    m_self[i] = 1;                       // masked until a node lands there
    m_cross[i] = (tok[i] == 0) ? 1 : 0;
  }
}

__global__ void bn_scatter_kernel(const float* __restrict__ xacc,
                                  const float* __restrict__ gb,
                                  const float* __restrict__ mu,
                                  const float* __restrict__ var,
                                  const float* __restrict__ g,
                                  const float* __restrict__ bb,
                                  const int* __restrict__ lab,
                                  const int* __restrict__ pos,
                                  int* m_self,
                                  float* xc, float* x1_out) {
  int i = blockIdx.x;
  int row = lab[i], p = pos[i];
  if (p >= cS) return;
  size_t o = ((size_t)row * cS + p) * cH;
  for (int c = threadIdx.x; c < cH; c += blockDim.x) {
    float v = xacc[(size_t)i * cH + c] + gb[c];
    v = (v - mu[c]) * rsqrtf(var[c] + 1e-5f) * g[c] + bb[c];
    v = fmaxf(v, 0.f);
    xc[o + c] = v;
    x1_out[o + c] = v;
  }
  if (threadIdx.x == 0) m_self[row * cS + p] = 0;
}

// ---------------------------------------------------------------------------
// LayerNorm (matches reference: ddof=1 variance, eps added to sqrt)
// ---------------------------------------------------------------------------
__global__ void ln_kernel(const float* __restrict__ in,
                          const float* __restrict__ delta,
                          float* __restrict__ out,
                          const float* __restrict__ a,
                          const float* __restrict__ b, int D) {
  __shared__ float sh[256];
  __shared__ float sm, sv;
  int row = blockIdx.x, t = threadIdx.x;
  size_t o = (size_t)row * D;
  float s = 0.f;
  for (int c = t; c < D; c += blockDim.x)
    s += in[o + c] + (delta ? delta[o + c] : 0.f);
  float tot = block_reduce_sum(s, sh);
  if (t == 0) sm = tot / D;
  __syncthreads();
  float m = sm;
  float s2 = 0.f;
  for (int c = t; c < D; c += blockDim.x) {
    float v = in[o + c] + (delta ? delta[o + c] : 0.f) - m;
    s2 += v * v;
  }
  float tot2 = block_reduce_sum(s2, sh);
  if (t == 0) sv = 1.f / (sqrtf(tot2 / (D - 1)) + 1e-6f);
  __syncthreads();
  float inv = sv;
  for (int c = t; c < D; c += blockDim.x) {
    float v = in[o + c] + (delta ? delta[o + c] : 0.f);
    out[o + c] = a[c] * (v - m) * inv + b[c];
  }
}

// ---------------------------------------------------------------------------
// attention pooling head
// ---------------------------------------------------------------------------
__global__ void att_logits_kernel(const float* __restrict__ t1,
                                  const float* __restrict__ w2,
                                  const float* __restrict__ b2,
                                  const int* __restrict__ m_self,
                                  float* attl, int D) {
  __shared__ float sh[256];
  int row = blockIdx.x, t = threadIdx.x;
  float s = 0.f;
  for (int c = t; c < D; c += 256) s += t1[(size_t)row * D + c] * w2[c];
  float tot = block_reduce_sum(s, sh);
  if (t == 0) attl[row] = m_self[row] ? -1.0e9f : (tot + b2[0]);
}

__global__ void att_softmax_kernel(float* attl) {
  __shared__ float sh[128];
  __shared__ float smax, ssum;
  int b = blockIdx.x, t = threadIdx.x;
  float v = attl[b * cS + t];
  sh[t] = v;
  __syncthreads();
  for (int s = 64; s > 0; s >>= 1) {
    if (t < s) sh[t] = fmaxf(sh[t], sh[t + s]);
    __syncthreads();
  }
  if (t == 0) smax = sh[0];
  __syncthreads();
  float ev = __expf(v - smax);
  sh[t] = ev;
  __syncthreads();
  for (int s = 64; s > 0; s >>= 1) {
    if (t < s) sh[t] += sh[t + s];
    __syncthreads();
  }
  if (t == 0) ssum = sh[0];
  __syncthreads();
  attl[b * cS + t] = ev / ssum;
}

__global__ void att_pool_kernel(const float* __restrict__ attl,
                                const float* __restrict__ xc,
                                float* pooled) {
  int b = blockIdx.x;
  for (int c = threadIdx.x; c < cH; c += blockDim.x) {
    float s = 0.f;
    for (int t = 0; t < cS; ++t)
      s += attl[b * cS + t] * xc[((size_t)b * cS + t) * cH + c];
    pooled[(size_t)b * cH + c] = s;
  }
}

// ---------------------------------------------------------------------------
// host
// ---------------------------------------------------------------------------
static void launch_gemm(const float* A, int lda, const _Float16* Bt,
                        const float* bias, float* C, int ldc,
                        int M, int N, int K, int act, hipStream_t st) {
  dim3 g((N + 63) / 64, (M + 63) / 64);
  wmma_gemm_kernel<<<g, dim3(256), 0, st>>>(A, lda, Bt, bias, C, ldc,
                                            M, N, K, act);
}

static void launch_wconv(const float* Wsrc, _Float16* Wt, int K, int N, int Npad,
                         hipStream_t st) {
  wconv_kernel<<<512, 256, 0, st>>>(Wsrc, Wt, K, N, Npad);
}

static void launch_zero(float* p, size_t n, hipStream_t st) {
  zero_kernel<<<1024, 256, 0, st>>>(p, n);
}

extern "C" void kernel_launch(void* const* d_in, const int* in_sizes, int n_in,
                              void* d_out, int out_size, void* d_ws, size_t ws_size,
                              hipStream_t stream) {
  (void)in_sizes; (void)n_in; (void)out_size; (void)ws_size;

  const float* nf       = (const float*)d_in[0];
  const int*   ei       = (const int*)  d_in[1];
  const float* ques     = (const float*)d_in[2];   // [B,1,S,H] contiguous
  const int*   lab      = (const int*)  d_in[3];
  const int*   tok      = (const int*)  d_in[4];   // [B,1,S]
  const float* gat_W    = (const float*)d_in[6];
  const float* gat_asrc = (const float*)d_in[7];
  const float* gat_adst = (const float*)d_in[8];
  const float* gat_b    = (const float*)d_in[9];
  const float* bn_g     = (const float*)d_in[10];
  const float* bn_b     = (const float*)d_in[11];
  const float* d1q_w = (const float*)d_in[12]; const float* d1q_b = (const float*)d_in[13];
  const float* d1k_w = (const float*)d_in[14]; const float* d1k_b = (const float*)d_in[15];
  const float* d1v_w = (const float*)d_in[16]; const float* d1v_b = (const float*)d_in[17];
  const float* d1m_w = (const float*)d_in[18]; const float* d1m_b = (const float*)d_in[19];
  const float* d2q_w = (const float*)d_in[20]; const float* d2q_b = (const float*)d_in[21];
  const float* d2k_w = (const float*)d_in[22]; const float* d2k_b = (const float*)d_in[23];
  const float* d2v_w = (const float*)d_in[24]; const float* d2v_b = (const float*)d_in[25];
  const float* d2m_w = (const float*)d_in[26]; const float* d2m_b = (const float*)d_in[27];
  const float* f1_w  = (const float*)d_in[28]; const float* f1_b  = (const float*)d_in[29];
  const float* f2_w  = (const float*)d_in[30]; const float* f2_b  = (const float*)d_in[31];
  const float* n2_a  = (const float*)d_in[32]; const float* n2_b  = (const float*)d_in[33];
  const float* n3_a  = (const float*)d_in[34]; const float* n3_b  = (const float*)d_in[35];
  const float* af1_w = (const float*)d_in[36]; const float* af1_b = (const float*)d_in[37];
  const float* af2_w = (const float*)d_in[38]; const float* af2_b = (const float*)d_in[39];
  const float* afm_w = (const float*)d_in[40]; const float* afm_b = (const float*)d_in[41];
  const float* pn_a  = (const float*)d_in[42]; const float* pn_b  = (const float*)d_in[43];
  const float* pj_w  = (const float*)d_in[44]; const float* pj_b  = (const float*)d_in[45];

  // output layout: out[B,NCLS] | x_t[B,FLAT] | x1[B,S,H] | alpha[EE]
  float* out0      = (float*)d_out;
  float* out_xt    = out0 + (size_t)cB * cNCLS;
  float* out_x1    = out_xt + (size_t)cB * cFLAT;
  float* out_alpha = out_x1 + (size_t)cBS * cH;

  // workspace carve (f32)
  float* W = (float*)d_ws;
  size_t off = 0;
  auto alloc = [&](size_t n) { float* p = W + off; off += n; return p; };
  float*    h      = alloc((size_t)cN * cH);
  float*    hsrc   = alloc(cN);
  float*    hdst   = alloc(cN);
  float*    e_buf  = alloc(cEE);
  float*    ee     = alloc(cEE);
  float*    den    = alloc(cN);
  unsigned* emax   = (unsigned*)alloc(cN);
  float*    xacc   = alloc((size_t)cN * cH);
  float*    mu     = alloc(cH);
  float*    var    = alloc(cH);
  int*      counts = (int*)alloc(cB);
  int*      offs   = (int*)alloc(cB);
  int*      pos    = (int*)alloc(cN);
  int*      mself  = (int*)alloc(cBS);
  int*      mcross = (int*)alloc(cBS);
  float*    alpha  = alloc(cEE);
  float*    xc     = alloc((size_t)cBS * cH);
  float*    qb     = alloc((size_t)cBS * cH);
  float*    kb     = alloc((size_t)cBS * cH);
  float*    vb     = alloc((size_t)cBS * cH);
  float*    ao     = alloc((size_t)cBS * cH);
  float*    t1     = alloc((size_t)cBS * cH);
  float*    ffb    = alloc((size_t)cBS * cFF);
  float*    pooled = alloc((size_t)cB * cH);
  float*    attl   = alloc((size_t)cBS);
  float*    xln    = alloc((size_t)cB * cFLAT);

  // workspace carve (f16 transposed weights), 16B-aligned
  off = (off + 7) & ~(size_t)7;
  _Float16* hbase = (_Float16*)(W + off);
  size_t offh = 0;
  auto hall = [&](size_t n) { _Float16* p = hbase + offh; offh += n; return p; };
  _Float16* gatWt = hall((size_t)cH * cIN);
  _Float16* wq1 = hall((size_t)cH * cH); _Float16* wk1 = hall((size_t)cH * cH);
  _Float16* wv1 = hall((size_t)cH * cH); _Float16* wm1 = hall((size_t)cH * cH);
  _Float16* wq2 = hall((size_t)cH * cH); _Float16* wk2 = hall((size_t)cH * cH);
  _Float16* wv2 = hall((size_t)cH * cH); _Float16* wm2 = hall((size_t)cH * cH);
  _Float16* wf1 = hall((size_t)cFF * cH);       // [2048][512]
  _Float16* wf2 = hall((size_t)cH * cFF);       // [512][2048]
  _Float16* af1t = hall((size_t)cH * cH);
  _Float16* afmt = hall((size_t)cFLAT * cH);
  _Float16* pjt  = hall((size_t)cNCLSP * cFLAT);

  // ---- GAT ----
  launch_wconv(gat_W, gatWt, cIN, cH, cH, stream);
  launch_gemm(nf, cIN, gatWt, nullptr, h, cH, cN, cH, cIN, 0, stream);
  rowdot2_kernel<<<cN, 256, 0, stream>>>(h, gat_asrc, gat_adst, hsrc, hdst, cH);

  launch_zero(den, cN, stream);
  launch_zero((float*)emax, cN, stream);          // 0 encodes below any real value
  launch_zero(xacc, (size_t)cN * cH, stream);
  launch_zero((float*)counts, cB, stream);

  int eb = (cEE + 255) / 256;
  edge_e_max_kernel  <<<eb, 256, 0, stream>>>(ei, hsrc, hdst, e_buf, emax);
  edge_softmax_kernel<<<eb, 256, 0, stream>>>(ei, e_buf, emax, ee, den);
  edge_alpha_kernel  <<<eb, 256, 0, stream>>>(ei, ee, den, alpha, out_alpha);
  scatter_feat_kernel<<<cEE, 256, 0, stream>>>(ei, h, alpha, xacc);
  bn_stats_kernel    <<<cH, 256, 0, stream>>>(xacc, gat_b, mu, var);

  count_labels_kernel<<<(cN + 255) / 256, 256, 0, stream>>>(lab, counts);
  scan_offs_kernel   <<<1, 32, 0, stream>>>(counts, offs);
  compute_pos_kernel <<<(cN + 255) / 256, 256, 0, stream>>>(lab, offs, pos);
  init_masks_kernel  <<<(cBS + 255) / 256, 256, 0, stream>>>(tok, mself, mcross);

  launch_zero(xc, (size_t)cBS * cH, stream);
  launch_zero(out_x1, (size_t)cBS * cH, stream);
  bn_scatter_kernel<<<cN, 256, 0, stream>>>(xacc, gat_b, mu, var, bn_g, bn_b,
                                            lab, pos, mself, xc, out_x1);

  // ---- decoder ----
  dim3 attg(cS / 32, cNH, cB);
  for (int i = 0; i < cL; ++i) {
    const size_t wo = (size_t)i * cH * cH, bo = (size_t)i * cH;
    // per-layer weight conversion (transposed f16)
    launch_wconv(d1q_w + wo, wq1, cH, cH, cH, stream);
    launch_wconv(d1k_w + wo, wk1, cH, cH, cH, stream);
    launch_wconv(d1v_w + wo, wv1, cH, cH, cH, stream);
    launch_wconv(d1m_w + wo, wm1, cH, cH, cH, stream);
    launch_wconv(d2q_w + wo, wq2, cH, cH, cH, stream);
    launch_wconv(d2k_w + wo, wk2, cH, cH, cH, stream);
    launch_wconv(d2v_w + wo, wv2, cH, cH, cH, stream);
    launch_wconv(d2m_w + wo, wm2, cH, cH, cH, stream);
    launch_wconv(f1_w + (size_t)i * cH * cFF, wf1, cH, cFF, cFF, stream);
    launch_wconv(f2_w + (size_t)i * cFF * cH, wf2, cFF, cH, cH, stream);
    // self-attention
    launch_gemm(xc, cH, wq1, d1q_b + bo, qb, cH, cBS, cH, cH, 0, stream);
    launch_gemm(xc, cH, wk1, d1k_b + bo, kb, cH, cBS, cH, cH, 0, stream);
    launch_gemm(xc, cH, wv1, d1v_b + bo, vb, cH, cBS, cH, cH, 0, stream);
    attention_kernel<<<attg, 256, 0, stream>>>(qb, kb, vb, mself, ao);
    launch_gemm(ao, cH, wm1, d1m_b + bo, t1, cH, cBS, cH, cH, 0, stream);
    ln_kernel<<<cBS, 256, 0, stream>>>(xc, t1, xc, n2_a + bo, n2_b + bo, cH);
    // cross-attention (K,V from question tokens)
    launch_gemm(xc,   cH, wq2, d2q_b + bo, qb, cH, cBS, cH, cH, 0, stream);
    launch_gemm(ques, cH, wk2, d2k_b + bo, kb, cH, cBS, cH, cH, 0, stream);
    launch_gemm(ques, cH, wv2, d2v_b + bo, vb, cH, cBS, cH, cH, 0, stream);
    attention_kernel<<<attg, 256, 0, stream>>>(qb, kb, vb, mcross, ao);
    launch_gemm(ao, cH, wm2, d2m_b + bo, t1, cH, cBS, cH, cH, 0, stream);
    ln_kernel<<<cBS, 256, 0, stream>>>(xc, t1, xc, n2_a + bo, n2_b + bo, cH);
    // FFN
    launch_gemm(xc,  cH,  wf1, f1_b + (size_t)i * cFF, ffb, cFF,
                cBS, cFF, cH, 1, stream);
    launch_gemm(ffb, cFF, wf2, f2_b + bo, t1, cH, cBS, cH, cFF, 0, stream);
    ln_kernel<<<cBS, 256, 0, stream>>>(xc, t1, xc, n3_a + bo, n3_b + bo, cH);
  }

  // ---- attention pooling + head ----
  launch_wconv(af1_w, af1t, cH, cH, cH, stream);
  launch_gemm(xc, cH, af1t, af1_b, t1, cH, cBS, cH, cH, 1, stream);
  att_logits_kernel <<<cBS, 256, 0, stream>>>(t1, af2_w, af2_b, mself, attl, cH);
  att_softmax_kernel<<<cB, 128, 0, stream>>>(attl);
  att_pool_kernel   <<<cB, 256, 0, stream>>>(attl, xc, pooled);

  launch_wconv(afm_w, afmt, cH, cFLAT, cFLAT, stream);
  launch_gemm(pooled, cH, afmt, afm_b, out_xt, cFLAT, cB, cFLAT, cH, 0, stream);
  ln_kernel<<<cB, 256, 0, stream>>>(out_xt, nullptr, xln, pn_a, pn_b, cFLAT);
  launch_wconv(pj_w, pjt, cFLAT, cNCLS, cNCLSP, stream);
  launch_gemm(xln, cFLAT, pjt, pj_b, out0, cNCLS, cB, cNCLS, cFLAT, 0, stream);
}
